// GATModel_37907381354602
// MI455X (gfx1250) — matrix operand order repaired and checked by reading, not compiled, not run on hardware
//
#include <hip/hip_runtime.h>
#include <hip/hip_bf16.h>
#include <math.h>

// ---------------------------------------------------------------------------
// GAT (2-layer, PyG-style) for MI455X / gfx1250.
// Bandwidth-bound edge scatter (fits in 192MB L2) + WMMA f16 GEMMs.
// ---------------------------------------------------------------------------

typedef __attribute__((ext_vector_type(16))) _Float16 v16h;
typedef __attribute__((ext_vector_type(8)))  _Float16 v8h;
typedef __attribute__((ext_vector_type(8)))  float    v8f;

#define NEG_SLOPE 0.2f

__device__ __forceinline__ float leaky(float x) { return x > 0.f ? x : NEG_SLOPE * x; }

// Monotonic-bits float atomic max: non-negative floats order as int,
// negative floats order reversed as unsigned. Init value must be -inf.
__device__ __forceinline__ void atomicMaxF(float* addr, float val) {
  if (val >= 0.f) atomicMax((int*)addr, __float_as_int(val));
  else            atomicMin((unsigned int*)addr, (unsigned int)__float_as_int(val));
}

// ---------------------------------------------------------------------------
// WMMA GEMM: Y[nrows, NC] = (X[nrows, K] (+ inbias[K])) @ W[K, NC]
// Block: 256 threads = 8 wave32 waves, each wave computes one 16x16 tile.
// X tile staged f32->f16 row-major; W staged f32->f16 TRANSPOSED (NC x K) so
// each lane's B fragment (k = kb + bko + h) is contiguous -> ds_load_b128.
// Row padding (+8 halfs, 272B stride) skews LDS banks (4 banks/row).
// ---------------------------------------------------------------------------
template<int K, int NC, int ROWT, int COLT>
__global__ __launch_bounds__(256) void gemm_wmma(
    const float* __restrict__ X, const float* __restrict__ W,
    const float* __restrict__ inbias, float* __restrict__ Y, int nrows)
{
  static_assert(ROWT * COLT == 8, "8 waves per block");
  static_assert(COLT * 16 == NC, "block spans all output columns");
  constexpr int ROWS = ROWT * 16;
  constexpr int PADH = 8;                 // +16B per row: 16B-aligned, bank skew
  constexpr int LDA  = K + PADH;          // sX row stride (halfs)
  constexpr int LDB  = K + PADH;          // sWT row stride (halfs)

  __shared__ alignas(16) _Float16 sX[ROWS * LDA];
  __shared__ alignas(16) _Float16 sWT[NC * LDB];

  const unsigned tid = threadIdx.x;
  const unsigned blockRow = blockIdx.x * ROWS;

  // Stage W transposed (f32 -> f16): sWT[col][k] = W[k][col].
  for (unsigned i = tid; i < K * NC; i += 256) {
    unsigned k = i / NC, c = i - k * NC;
    sWT[c * LDB + k] = (_Float16)W[i];
  }

  // Stage X tile (f32 -> f16), optional input bias fold (layer-1 bias).
  for (unsigned i = tid; i < ROWS * K; i += 256) {
    unsigned r = i / K, k = i - r * K;
    unsigned rr = blockRow + r;
    float v = 0.f;
    if (rr < (unsigned)nrows) {
      v = X[(size_t)rr * K + k];
      if (inbias) v += inbias[k];
    }
    sX[r * LDA + k] = (_Float16)v;
  }
  __syncthreads();

  const unsigned wave = tid >> 5;
  const unsigned lane = tid & 31;
  const unsigned rt = wave / COLT;          // row tile of this wave
  const unsigned ct = wave - rt * COLT;     // col tile of this wave

  v8f acc = {};

  // A-fragment (16x32 f16): lanes 0-15 rows M=0..15 K=0..15,
  //                         lanes 16-31 same rows K=16..31.
  const unsigned arow = rt * 16 + (lane & 15);
  const unsigned alo  = (lane >> 4) * 16;
  // B-fragment (32x16 f16): col = lane&15; lanes 0-15 K=0..15, 16-31 K=16..31.
  const unsigned bcol = ct * 16 + (lane & 15);
  const unsigned bko  = (lane >> 4) * 16;

  const _Float16* ap = &sX[arow * LDA];
  const _Float16* bp = &sWT[bcol * LDB];

  #pragma unroll
  for (int kb = 0; kb < K; kb += 32) {
    // All offsets are multiples of 8 halfs == 16 bytes -> aligned v8h loads.
    v8h aLo = *(const v8h*)(ap + kb + alo);        // VGPRs 0-3: k = kb+alo+0..7
    v8h aHi = *(const v8h*)(ap + kb + alo + 8);    // VGPRs 4-7: k = kb+alo+8..15
    v8h bLo = *(const v8h*)(bp + kb + bko);        // k = kb+bko+0..7
    v8h bHi = *(const v8h*)(bp + kb + bko + 8);    // k = kb+bko+8..15
    v16h a = __builtin_shufflevector(aLo, aHi, 0,1,2,3,4,5,6,7,8,9,10,11,12,13,14,15);
    v16h b = __builtin_shufflevector(bLo, bHi, 0,1,2,3,4,5,6,7,8,9,10,11,12,13,14,15);
    // D = A*B + C  (f32 accumulate) -> v_wmma_f32_16x16x32_f16
    acc = __builtin_amdgcn_wmma_f32_16x16x32_f16(
        /*neg_a=*/false, a, /*neg_b=*/false, b,
        /*c_mod=*/(short)0, acc, /*reuse_a=*/false, /*reuse_b=*/false);
  }

  // D layout: VGPR r -> row M = r (+8 for lanes 16-31), col = lane&15.
  const unsigned crow0 = blockRow + rt * 16 + ((lane >> 4) * 8);
  const unsigned ccol  = ct * 16 + (lane & 15);
  float* yp = Y + (size_t)crow0 * NC + ccol;
  if (crow0 + 7 < (unsigned)nrows) {
    // Fast path: 8 straight stores, constant byte offsets (fit 24-bit IOFFSET).
    #pragma unroll
    for (int r = 0; r < 8; ++r) yp[r * NC] = acc[r];
  } else {
    #pragma unroll
    for (int r = 0; r < 8; ++r)
      if (crow0 + r < (unsigned)nrows) yp[r * NC] = acc[r];
  }
}

// ---------------------------------------------------------------------------
// Per (node, head): attention logits al_src/al_dst, init m/-inf, den/0,
// and zero the aggregation buffer slice.
// ---------------------------------------------------------------------------
template<int C, int H>
__global__ __launch_bounds__(256) void attn_prep(
    const float* __restrict__ hfeat, const float* __restrict__ a_src,
    const float* __restrict__ a_dst, float* __restrict__ als,
    float* __restrict__ ald, float* __restrict__ m, float* __restrict__ den,
    float* __restrict__ agg, int n)
{
  unsigned i = blockIdx.x * blockDim.x + threadIdx.x;   // node*H + head
  if (i >= (unsigned)(n * H)) return;
  unsigned node = i / H, head = i - node * H;
  const float* hp = hfeat + (size_t)node * (H * C) + head * C;
  const float* as = a_src + head * C;
  const float* ad = a_dst + head * C;
  float s = 0.f, d = 0.f;
  #pragma unroll
  for (int c = 0; c < C; ++c) { float v = hp[c]; s += v * as[c]; d += v * ad[c]; }
  als[i] = s;
  ald[i] = d;
  m[i]   = -__builtin_inff();
  den[i] = 0.f;
  float* ag = agg + (size_t)node * (H * C) + head * C;
  #pragma unroll
  for (int c = 0; c < C; ++c) ag[c] = 0.f;
}

// ---------------------------------------------------------------------------
// Pass 1: segment max of leaky_relu(al_src[s]+al_dst[d]) over incoming edges.
// Edges [0,E) from edge_index; edges [E, E+n) are self loops.
// ---------------------------------------------------------------------------
template<int H>
__global__ __launch_bounds__(256) void edge_max(
    const int* __restrict__ src, const int* __restrict__ dst, int E, int n,
    const float* __restrict__ als, const float* __restrict__ ald,
    float* __restrict__ m)
{
  int e = blockIdx.x * blockDim.x + threadIdx.x;
  if (e >= E + n) return;
  int s = (e < E) ? src[e] : (e - E);
  int d = (e < E) ? dst[e] : (e - E);
  #pragma unroll
  for (int h = 0; h < H; ++h) {
    float v = leaky(als[s * H + h] + ald[d * H + h]);
    atomicMaxF(&m[d * H + h], v);
  }
}

// Pass 2: denom[d,h] += exp(e - m[d,h])
template<int H>
__global__ __launch_bounds__(256) void edge_expsum(
    const int* __restrict__ src, const int* __restrict__ dst, int E, int n,
    const float* __restrict__ als, const float* __restrict__ ald,
    const float* __restrict__ m, float* __restrict__ den)
{
  int e = blockIdx.x * blockDim.x + threadIdx.x;
  if (e >= E + n) return;
  int s = (e < E) ? src[e] : (e - E);
  int d = (e < E) ? dst[e] : (e - E);
  #pragma unroll
  for (int h = 0; h < H; ++h) {
    float v = leaky(als[s * H + h] + ald[d * H + h]);
    atomicAdd(&den[d * H + h], __expf(v - m[d * H + h]));
  }
}

// Pass 3: agg[d,h,:] += alpha * hfeat[s,h,:]  (one thread per edge-head)
// Source row gathered via float4 loads into registers, then scatter atomics.
template<int H, int C>
__global__ __launch_bounds__(256) void edge_agg(
    const int* __restrict__ src, const int* __restrict__ dst, int E, int n,
    const float* __restrict__ als, const float* __restrict__ ald,
    const float* __restrict__ m, const float* __restrict__ den,
    const float* __restrict__ hfeat, float* __restrict__ agg)
{
  int idx = blockIdx.x * blockDim.x + threadIdx.x;
  if (idx >= (E + n) * H) return;
  int e = idx / H, h = idx - e * H;

  // Speculative prefetch of the streamed edge-index arrays (safe: dropped on
  // translation miss per CDNA5 prefetch semantics).
  if (e + 8192 < E) {
    __builtin_prefetch(&src[e + 8192], 0, 1);
    __builtin_prefetch(&dst[e + 8192], 0, 1);
  }

  int s = (e < E) ? src[e] : (e - E);
  int d = (e < E) ? dst[e] : (e - E);
  float v = leaky(als[s * H + h] + ald[d * H + h]);
  float alpha = __expf(v - m[d * H + h]) / den[d * H + h];

  const float4* hv = (const float4*)(hfeat + (size_t)s * (H * C) + h * C);
  float* ag = agg + (size_t)d * (H * C) + h * C;
  #pragma unroll
  for (int q = 0; q < C / 4; ++q) {
    float4 f = hv[q];                      // global_load_b128 gather
    atomicAdd(&ag[4 * q + 0], f.x * alpha);
    atomicAdd(&ag[4 * q + 1], f.y * alpha);
    atomicAdd(&ag[4 * q + 2], f.z * alpha);
    atomicAdd(&ag[4 * q + 3], f.w * alpha);
  }
}

// ---------------------------------------------------------------------------
// Finalize: mean over 4 heads of agg2[N,4,32] + b2, ELU, log_softmax(axis=1).
// One wave32 per node: lane == output channel (32 channels, perfect fit).
// ---------------------------------------------------------------------------
__global__ __launch_bounds__(256) void finalize_k(
    const float* __restrict__ agg2, const float* __restrict__ b2,
    float* __restrict__ out, int n)
{
  unsigned node = blockIdx.x * 8 + (threadIdx.x >> 5);
  unsigned c = threadIdx.x & 31;
  if (node >= (unsigned)n) return;
  const float* a = agg2 + (size_t)node * 128;
  float v = 0.25f * (a[c] + a[32 + c] + a[64 + c] + a[96 + c]) + b2[c];
  v = (v > 0.f) ? v : (__expf(v) - 1.f);                 // ELU
  float mx = v;
  #pragma unroll
  for (int o = 16; o > 0; o >>= 1) mx = fmaxf(mx, __shfl_xor(mx, o, 32));
  float ex = __expf(v - mx);
  float sm = ex;
  #pragma unroll
  for (int o = 16; o > 0; o >>= 1) sm += __shfl_xor(sm, o, 32);
  out[(size_t)node * 32 + c] = v - mx - __logf(sm);
}

// ---------------------------------------------------------------------------
extern "C" void kernel_launch(void* const* d_in, const int* in_sizes, int n_in,
                              void* d_out, int out_size, void* d_ws, size_t ws_size,
                              hipStream_t stream) {
  const float* x      = (const float*)d_in[0];
  const int*   eidx   = (const int*)d_in[1];   // [2,E] (src row, dst row)
  const float* W1     = (const float*)d_in[2]; // [128,64]
  const float* a_src1 = (const float*)d_in[3]; // [4,16]
  const float* a_dst1 = (const float*)d_in[4];
  const float* b1     = (const float*)d_in[5]; // [64]
  const float* W2     = (const float*)d_in[6]; // [64,128]
  const float* a_src2 = (const float*)d_in[7]; // [4,32]
  const float* a_dst2 = (const float*)d_in[8];
  const float* b2     = (const float*)d_in[9]; // [32]

  const int N = in_sizes[0] / 128;
  const int E = in_sizes[1] / 2;
  const int Etot = E + N;
  const int* src = eidx;
  const int* dst = eidx + E;

  // Workspace carve-up (floats).
  float* ws = (float*)d_ws;
  float* h1   = ws;  ws += (size_t)N * 64;
  float* als1 = ws;  ws += (size_t)N * 4;
  float* ald1 = ws;  ws += (size_t)N * 4;
  float* m1   = ws;  ws += (size_t)N * 4;
  float* den1 = ws;  ws += (size_t)N * 4;
  float* agg1 = ws;  ws += (size_t)N * 64;
  float* h2   = ws;  ws += (size_t)N * 128;
  float* als2 = ws;  ws += (size_t)N * 4;
  float* ald2 = ws;  ws += (size_t)N * 4;
  float* m2   = ws;  ws += (size_t)N * 4;
  float* den2 = ws;  ws += (size_t)N * 4;
  float* agg2 = ws;  ws += (size_t)N * 128;

  const int TB = 256;
  const int gNH  = (N * 4 + TB - 1) / TB;
  const int gE   = (Etot + TB - 1) / TB;
  const int gEH  = (Etot * 4 + TB - 1) / TB;

  // ---- Layer 1: h1 = x @ W1 (WMMA), attention, softmax-scatter ----
  gemm_wmma<128, 64, 2, 4><<<(N + 31) / 32, TB, 0, stream>>>(x, W1, nullptr, h1, N);
  attn_prep<16, 4><<<gNH, TB, 0, stream>>>(h1, a_src1, a_dst1, als1, ald1, m1, den1, agg1, N);
  edge_max<4><<<gE, TB, 0, stream>>>(src, dst, E, N, als1, ald1, m1);
  edge_expsum<4><<<gE, TB, 0, stream>>>(src, dst, E, N, als1, ald1, m1, den1);
  edge_agg<4, 16><<<gEH, TB, 0, stream>>>(src, dst, E, N, als1, ald1, m1, den1, h1, agg1);

  // ---- Layer 2: h2 = (agg1 + b1) @ W2 (WMMA, bias folded into load) ----
  gemm_wmma<64, 128, 1, 8><<<(N + 15) / 16, TB, 0, stream>>>(agg1, W2, b1, h2, N);
  attn_prep<32, 4><<<gNH, TB, 0, stream>>>(h2, a_src2, a_dst2, als2, ald2, m2, den2, agg2, N);
  edge_max<4><<<gE, TB, 0, stream>>>(src, dst, E, N, als2, ald2, m2);
  edge_expsum<4><<<gE, TB, 0, stream>>>(src, dst, E, N, als2, ald2, m2, den2);
  edge_agg<4, 32><<<gEH, TB, 0, stream>>>(src, dst, E, N, als2, ald2, m2, den2, h2, agg2);

  // ---- Head-mean + b2 + ELU + log_softmax ----
  finalize_k<<<(N + 7) / 8, TB, 0, stream>>>(agg2, b2, (float*)d_out, N);
}
